// VectorQuantizer_1151051235449
// MI455X (gfx1250) — matrix operand order repaired and checked by reading, not compiled, not run on hardware
//
#include <hip/hip_runtime.h>
#include <hip/hip_bf16.h>

typedef __attribute__((ext_vector_type(16))) __bf16       v16bf;
typedef __attribute__((ext_vector_type(8)))  float        v8f;
typedef __attribute__((ext_vector_type(4)))  float        v4f;
typedef __attribute__((ext_vector_type(2)))  float        v2f;
typedef __attribute__((ext_vector_type(4)))  unsigned int v4u;

#define K_DIM        64
#define NUM_E        1024
#define N_TILES      (NUM_E / 16)                // 64
#define CB_STRIDE    72           // bf16 elems per LDS row (64 + 8 pad -> bank spread)
#define WG_THREADS   256
#define WAVES_PER_WG 8
#define GRID_WGS     512
#define TOTAL_WAVES  (GRID_WGS * WAVES_PER_WG)   // 4096
#define N_ROWS       262144                      // 64^4 / 64
#define M_TILES      (N_ROWS / 16)               // 16384
#define TILES_PER_WAVE (M_TILES / TOTAL_WAVES)   // 4

// f32 -> bf16 bits, round-to-nearest-even
static __device__ __forceinline__ unsigned short f2bf(float f) {
  union { float f; unsigned u; } cv; cv.f = f;
  unsigned r = cv.u + 0x7FFFu + ((cv.u >> 16) & 1u);
  return (unsigned short)(r >> 16);
}

union BU { v4u q[2]; v16bf v; };

__global__ void vq_init_loss(float* loss) { *loss = 0.0f; }

__global__ __launch_bounds__(WG_THREADS)
void vq_kernel(const float* __restrict__ z, const float* __restrict__ cb,
               float* __restrict__ out, float* __restrict__ loss) {
  // Whole codebook resident in LDS: 144 KB bf16 (padded) + 4 KB half-norms.
  // 148 KB/WG -> 2 workgroups (16 waves) per 320 KB WGP.
  __shared__ __align__(16) unsigned short sCB[NUM_E * CB_STRIDE];
  __shared__ float sNorm[NUM_E];

  const int tid   = threadIdx.x;
  const int lane  = tid & 31;
  const int wave  = tid >> 5;
  const int gwave = blockIdx.x * WAVES_PER_WG + wave;

  // ---- one-time staging: f32 codebook -> bf16 LDS (b128 loads, b64 DS stores) ----
#pragma unroll
  for (int j = 0; j < (NUM_E * K_DIM) / (4 * WG_THREADS); ++j) { // 64 iters
    const int f4   = tid + j * WG_THREADS;
    const int base = f4 * 4;
    const int e    = base >> 6;
    const int k    = base & 63;
    v4f d = ((const v4f*)cb)[f4];
    union { unsigned short s[4]; unsigned long long u; } pk;
    pk.s[0] = f2bf(d.x); pk.s[1] = f2bf(d.y); pk.s[2] = f2bf(d.z); pk.s[3] = f2bf(d.w);
    *(unsigned long long*)&sCB[e * CB_STRIDE + k] = pk.u;
  }
  // half squared norms (exact f32, matches reference distance bias)
  for (int e = tid; e < NUM_E; e += WG_THREADS) {
    const v4f* row = (const v4f*)(cb + (long)e * K_DIM);
    float s = 0.f;
#pragma unroll
    for (int q = 0; q < K_DIM / 4; ++q) {
      v4f d = row[q];
      s += d.x * d.x + d.y * d.y + d.z * d.z + d.w * d.w;
    }
    sNorm[e] = 0.5f * s;
  }
  __syncthreads();   // only barrier in the kernel: waves run free afterwards

  float lsum = 0.f;
  const int ecolL = lane & 15;
  const int kHalf = (lane >> 4) * 16;

  // B-tile fetch: lane L holds column N=L&15; elems 0..15 = K koff..koff+15,
  // koff = (L>>4)*16 + 32*c -> 32 contiguous bytes from the padded LDS row.
  auto loadB = [&](BU* b, float& hn, int nb) {
#pragma unroll
    for (int c = 0; c < 2; ++c) {
      const v4u* bp = (const v4u*)&sCB[(nb * 16 + ecolL) * CB_STRIDE + kHalf + c * 32];
      b[c].q[0] = bp[0];
      b[c].q[1] = bp[1];
    }
    hn = sNorm[nb * 16 + ecolL];
  };

  for (int t = 0; t < TILES_PER_WAVE; ++t) {
    const int  tile    = gwave + t * TOTAL_WAVES;
    const long rowBase = (long)tile * 16;

    // ---- Load A (16 rows x 64, bf16) in contiguous k order, same as B ----
    // lane L<16: row L, elems 0..15 = K kb..kb+15 (kb=0)
    // lane L>=16: same row, kb=16.
    const long arow = rowBase + (lane & 15);
    const int  kb   = (lane >> 4) * 16;
    v16bf A[2];
#pragma unroll
    for (int c = 0; c < 2; ++c) {
      const float* zp = z + arow * K_DIM + c * 32;
      union { v4f v[2]; float f[8]; } lo, hi;
      lo.v[0] = ((const v4f*)(zp + kb))[0];
      lo.v[1] = ((const v4f*)(zp + kb))[1];
      hi.v[0] = ((const v4f*)(zp + kb + 8))[0];
      hi.v[1] = ((const v4f*)(zp + kb + 8))[1];
      union { unsigned short s[16]; v16bf v; } ua;
#pragma unroll
      for (int j = 0; j < 8; ++j) { ua.s[j] = f2bf(lo.f[j]); ua.s[8 + j] = f2bf(hi.f[j]); }
      A[c] = ua.v;
    }

    float    best[8];
    unsigned bidx[8];
#pragma unroll
    for (int v = 0; v < 8; ++v) { best[v] = -3.4e38f; bidx[v] = 0u; }

    auto compute = [&](const BU* b, float hn, int nbase) {
      v8f acc = {};
      acc = __builtin_amdgcn_wmma_f32_16x16x32_bf16(
          false, A[0], false, b[0].v, (short)0, acc, false, false);
      acc = __builtin_amdgcn_wmma_f32_16x16x32_bf16(
          false, A[1], false, b[1].v, (short)0, acc, false, false);
      const unsigned n = (unsigned)(nbase + ecolL);
#pragma unroll
      for (int v = 0; v < 8; ++v) {
        float s = acc[v] - hn;                 // z.e - 0.5*||e||^2 ; argmax == argmin dist
        if (s > best[v]) { best[v] = s; bidx[v] = n; }
      }
    };

    // ---- software-pipelined sweep: prefetch B(nb+1) while computing B(nb) ----
    BU  bEven[2], bOdd[2];
    float hEven, hOdd;
    loadB(bEven, hEven, 0);
    for (int nb = 0; nb < N_TILES; nb += 2) {
      loadB(bOdd, hOdd, nb + 1);               // issue ds_loads an iteration ahead
      compute(bEven, hEven, nb * 16);
      const int nb2 = (nb + 2 < N_TILES) ? nb + 2 : (N_TILES - 1);
      loadB(bEven, hEven, nb2);                // (clamped redundant load at tail)
      compute(bOdd, hOdd, (nb + 1) * 16);
    }

    // ---- branchless cross-lane argmax within each 16-lane half ----
#pragma unroll
    for (int off = 1; off < 16; off <<= 1) {
#pragma unroll
      for (int v = 0; v < 8; ++v) {
        float    ob   = __shfl_xor(best[v], off, 32);
        unsigned oi   = __shfl_xor(bidx[v], off, 32);
        bool     take = (ob > best[v]) | ((ob == best[v]) & (oi < bidx[v]));
        best[v] = take ? ob : best[v];
        bidx[v] = take ? oi : bidx[v];
      }
    }

    // ---- gather winning codebook rows (f32, L2-resident), write out, loss ----
#pragma unroll
    for (int r = 0; r < 16; ++r) {
      const unsigned e = (unsigned)__shfl((int)bidx[r & 7], (r < 8) ? 0 : 16, 32);
      const v2f* qp = (const v2f*)(cb + (long)e * K_DIM);
      const v2f* zp = (const v2f*)(z + (rowBase + r) * K_DIM);
      v2f q  = qp[lane];
      v2f zz = zp[lane];
      ((v2f*)(out + (rowBase + r) * K_DIM))[lane] = q;
      float dx = q.x - zz.x, dy = q.y - zz.y;
      lsum += dx * dx + dy * dy;
    }
  }

  // loss = q_loss + 0.25*e_loss = 1.25 * sum/(64^4)
#pragma unroll
  for (int off = 1; off < 32; off <<= 1) lsum += __shfl_xor(lsum, off, 32);
  if (lane == 0) atomicAdd(loss, lsum * (1.25f / 16777216.0f));
}

extern "C" void kernel_launch(void* const* d_in, const int* in_sizes, int n_in,
                              void* d_out, int out_size, void* d_ws, size_t ws_size,
                              hipStream_t stream) {
  (void)in_sizes; (void)n_in; (void)d_ws; (void)ws_size; (void)out_size;
  const float* z  = (const float*)d_in[0];   // (64,64,64,64) f32
  const float* cb = (const float*)d_in[1];   // (1024,64) f32
  float* out  = (float*)d_out;               // quantized_st, 16777216 f32
  float* loss = out + 16777216;              // scalar loss slot

  vq_init_loss<<<1, 1, 0, stream>>>(loss);
  vq_kernel<<<GRID_WGS, WG_THREADS, 0, stream>>>(z, cb, out, loss);
}
